// HetNetGNN_v3_50044958933535
// MI455X (gfx1250) — matrix operand order repaired and checked by reading, not compiled, not run on hardware
//
#include <hip/hip_runtime.h>
#include <math.h>

#define N_UE 4096
#define N_AP 128
#define CDIV(a,b) (((a)+(b)-1)/(b))

typedef __attribute__((ext_vector_type(16))) __bf16 v16bf;
typedef __attribute__((ext_vector_type(8)))  float v8f;

// ---------------------------------------------------------------------------
// Edge message + aggregate kernel (WMMA bf16, f32 accumulate).
//   h1[16]  = relu(a_src[src] + a_dst[dst] + e*w1e + b1)
//   msg[C]  = relu(h1 @ W2 + b2)              (b2 folded into WMMA C operand)
//   out[dst,:] += sum over srcs of msg        (atomic f32 add)
// Edge scalars pre-transposed: et[dst*N_src + src] (coalesced).
// One wave per (dst, split); 16 edges per WMMA tile; 2-deep A double-buffer
// (persistent zero K-pad, no WAR hazard, no in-loop branch).
// ---------------------------------------------------------------------------
__global__ __launch_bounds__(32) void k_edge_msg(
    const float* __restrict__ et,      // [N_dst, N_src] packed edge col0
    const float* __restrict__ a_src,   // [N_src,16] or nullptr
    const float* __restrict__ a_dst,   // [N_dst,16] or nullptr
    const float* __restrict__ w1e,     // [16]
    const float* __restrict__ b1,      // [16]
    const float* __restrict__ W2,      // [16, Cout] row-major
    const float* __restrict__ b2,      // [Cout]
    float* __restrict__ out,           // [N_dst, ldo] atomic-accumulated
    int Cout, int ldo, int N_src, int srcPerSplit)
{
  const int lane = threadIdx.x;       // 0..31
  const int m    = lane & 15;         // A-matrix row handled by this lane
  const int kb   = (lane >> 4) * 8;   // K-chunk base (0 or 8)
  const int dst  = blockIdx.x;
  const int s0   = blockIdx.y * srcPerSplit;

  // B = W2 padded to K=32 (rows kb..kb+7 in elements 0..7; elements 8..15 zero). Two N-halves.
  v16bf Blo{}, Bhi{};
  {
#pragma unroll
    for (int j = 0; j < 8; ++j) {
      Blo[j] = (__bf16)W2[(kb + j) * Cout + m];
      int n2 = 16 + m;
      Bhi[j] = (__bf16)((n2 < Cout) ? W2[(kb + j) * Cout + n2] : 0.0f);
    }
  }

  float w1k[8], b1k[8], adk[8];
#pragma unroll
  for (int j = 0; j < 8; ++j) { w1k[j] = w1e[kb + j]; b1k[j] = b1[kb + j]; }
  if (a_dst) {
#pragma unroll
    for (int j = 0; j < 8; ++j) adk[j] = a_dst[(size_t)dst * 16 + kb + j];
  } else {
#pragma unroll
    for (int j = 0; j < 8; ++j) adk[j] = 0.0f;
  }

  // Fold output bias into the WMMA C operand: C[m][n] = b2[n] (lane = n).
  const float b2lo = b2[m];
  const float b2hi = (16 + m < Cout) ? b2[16 + m] : 0.0f;
  v8f Clo, Chi;
#pragma unroll
  for (int i = 0; i < 8; ++i) { Clo[i] = b2lo; Chi[i] = b2hi; }

  const float* erow = et + (size_t)dst * N_src + s0;  // coalesced edge row

  // Build A-tile t: writes only elements 0..7; 8..15 are persistent K-pad 0s.
  auto mkA = [&](int t, v16bf& A) {
    const int so = t * 16 + m;
    const float e = erow[so];
    float h[8];
    if (a_src) {
      const float4* p =
          reinterpret_cast<const float4*>(a_src + (size_t)(s0 + so) * 16 + kb);
      float4 x0 = p[0], x1 = p[1];
      h[0]=x0.x; h[1]=x0.y; h[2]=x0.z; h[3]=x0.w;
      h[4]=x1.x; h[5]=x1.y; h[6]=x1.z; h[7]=x1.w;
    } else {
#pragma unroll
      for (int j = 0; j < 8; ++j) h[j] = 0.0f;
    }
#pragma unroll
    for (int j = 0; j < 8; ++j) {
      float v = fmaf(e, w1k[j], h[j] + adk[j] + b1k[j]);
      A[j] = (__bf16)(v > 0.0f ? v : 0.0f);
    }
  };

  v8f accLo{}, accHi{};
  auto step = [&](const v16bf& A) {
    v8f dlo = __builtin_amdgcn_wmma_f32_16x16x32_bf16(false, A, false, Blo,
                                                      (short)0, Clo, false, false);
    v8f dhi = __builtin_amdgcn_wmma_f32_16x16x32_bf16(false, A, false, Bhi,
                                                      (short)0, Chi, false, false);
#pragma unroll
    for (int i = 0; i < 8; ++i) {
      accLo[i] += (dlo[i] > 0.0f ? dlo[i] : 0.0f);
      accHi[i] += (dhi[i] > 0.0f ? dhi[i] : 0.0f);
    }
  };

  const int ntiles = srcPerSplit >> 4;   // always even (>= 2)
  v16bf A0{}, A1{};                      // K-pad halves zeroed once, kept
  mkA(0, A0);
  int t = 0;
  for (; t < ntiles - 2; t += 2) {
    mkA(t + 1, A1);                      // independent work in WMMA shadow
    step(A0);
    mkA(t + 2, A0);
    step(A1);
  }
  mkA(t + 1, A1);
  step(A0);
  step(A1);

  // Reduce over the 16 edge rows: 8 VGPRs per lane + cross-half shuffle.
  float sLo = 0.0f, sHi = 0.0f;
#pragma unroll
  for (int i = 0; i < 8; ++i) { sLo += accLo[i]; sHi += accHi[i]; }
  sLo += __shfl_xor(sLo, 16, 32);
  sHi += __shfl_xor(sHi, 16, 32);
  if (lane < 16) {
    atomicAdd(out + (size_t)dst * ldo + lane, sLo);
    int n2 = 16 + lane;
    if (n2 < Cout) atomicAdd(out + (size_t)dst * ldo + n2, sHi);
  }
}

// T[d*N_src + s] = E[(s*N_dst + d)*2]  (pack col0 + transpose, one-time)
__global__ void k_pack_transpose(const float* __restrict__ E, float* __restrict__ T,
                                 int N_src, int N_dst) {
  int i = blockIdx.x * blockDim.x + threadIdx.x;
  if (i >= N_src * N_dst) return;
  int d = i / N_src, s = i - d * N_src;
  T[i] = E[((size_t)s * N_dst + d) * 2];
}

// out16[n,k] = sum_c x[n,c] * W1[c,k]   (per-node hoisted first-layer partial)
__global__ void k_lin_partial(const float* __restrict__ x, int ldx, int Cin,
                              const float* __restrict__ W1,
                              float* __restrict__ out16, int N) {
  int idx = blockIdx.x * blockDim.x + threadIdx.x;
  int n = idx >> 4, k = idx & 15;
  if (n >= N) return;
  float s = 0.0f;
  for (int c = 0; c < Cin; ++c) s = fmaf(x[(size_t)n * ldx + c], W1[c * 16 + k], s);
  out16[(size_t)n * 16 + k] = s;
}

// out[n,:Cout] = relu(relu(x[n,:Cin]@W1+b1)@W2+b2)
__global__ void k_node_mlp2(const float* __restrict__ x, int ldx, int Cin,
                            const float* __restrict__ W1, const float* __restrict__ b1,
                            const float* __restrict__ W2, const float* __restrict__ b2,
                            int Cout, float* __restrict__ out, int ldo, int N) {
  int n = blockIdx.x * blockDim.x + threadIdx.x;
  if (n >= N) return;
  float h[16];
#pragma unroll
  for (int k = 0; k < 16; ++k) h[k] = b1[k];
  for (int c = 0; c < Cin; ++c) {
    float xv = x[(size_t)n * ldx + c];
#pragma unroll
    for (int k = 0; k < 16; ++k) h[k] = fmaf(xv, W1[c * 16 + k], h[k]);
  }
#pragma unroll
  for (int k = 0; k < 16; ++k) h[k] = h[k] > 0.0f ? h[k] : 0.0f;
  for (int co = 0; co < Cout; ++co) {
    float s = b2[co];
#pragma unroll
    for (int k = 0; k < 16; ++k) s = fmaf(h[k], W2[k * Cout + co], s);
    out[(size_t)n * ldo + co] = s > 0.0f ? s : 0.0f;
  }
}

__global__ void k_colsum(const float* __restrict__ t, int N, int C, float* __restrict__ S) {
  __shared__ float red[256];
  int c = blockIdx.x;
  float s = 0.0f;
  for (int n = threadIdx.x; n < N; n += blockDim.x) s += t[(size_t)n * C + c];
  red[threadIdx.x] = s; __syncthreads();
  for (int off = 128; off > 0; off >>= 1) {
    if ((int)threadIdx.x < off) red[threadIdx.x] += red[threadIdx.x + off];
    __syncthreads();
  }
  if (threadIdx.x == 0) S[c] = red[0];
}

__global__ void k_bcast(float* __restrict__ out, const float* __restrict__ S, int total, int C) {
  int i = blockIdx.x * blockDim.x + threadIdx.x;
  if (i < total) out[i] = S[i % C];
}

__global__ void k_copy_strided(float* __restrict__ dst, int ds,
                               const float* __restrict__ src, int ss, int N) {
  int i = blockIdx.x * blockDim.x + threadIdx.x;
  if (i < N) dst[(size_t)i * ds] = src[(size_t)i * ss];
}

__global__ void k_copy(float* __restrict__ dst, const float* __restrict__ src, int N) {
  int i = blockIdx.x * blockDim.x + threadIdx.x;
  if (i < N) dst[i] = src[i];
}

// power head: out[2n] = xu[n,0]; out[2n+1] = sigmoid(relu(xu@W1+b1)@W2+b2)
__global__ void k_power(const float* __restrict__ xu,
                        const float* __restrict__ W1, const float* __restrict__ b1,
                        const float* __restrict__ W2, const float* __restrict__ b2,
                        float* __restrict__ out, int N) {
  int n = blockIdx.x * blockDim.x + threadIdx.x;
  if (n >= N) return;
  float h[16];
#pragma unroll
  for (int k = 0; k < 16; ++k) h[k] = b1[k];
  for (int c = 0; c < 32; ++c) {
    float xv = xu[(size_t)n * 32 + c];
#pragma unroll
    for (int k = 0; k < 16; ++k) h[k] = fmaf(xv, W1[c * 16 + k], h[k]);
  }
  float z = b2[0];
#pragma unroll
  for (int k = 0; k < 16; ++k) z = fmaf(h[k] > 0.0f ? h[k] : 0.0f, W2[k], z);
  out[2 * n]     = xu[(size_t)n * 32];
  out[2 * n + 1] = 1.0f / (1.0f + __expf(-z));
}

extern "C" void kernel_launch(void* const* d_in, const int* in_sizes, int n_in,
                              void* d_out, int out_size, void* d_ws, size_t ws_size,
                              hipStream_t stream) {
  (void)in_sizes; (void)n_in; (void)out_size; (void)ws_size;
  const float* x_ue = (const float*)d_in[0];
  const float* eup  = (const float*)d_in[3];
  const float* edn  = (const float*)d_in[4];
#define PF(i) ((const float*)d_in[(i)])
  enum { c1_msg_ue = 5, c1_edge_up = 9, c1_edge_dn = 13, c1_upd_ue = 17,
         c2_msg_ue = 21, c2_msg_ap = 25, c2_upd_ue = 29, c2_upd_ap = 33,
         c3_msg_ue = 37, c3_msg_ap = 41, c3_upd_ue = 45, c3_upd_ap = 49,
         power1 = 53, power2 = 55 };

  float* ws   = (float*)d_ws;
  float* xu0  = ws;                   // [4096,32]
  float* xu1  = ws + 131072;          // [4096,32]
  float* xa0  = ws + 262144;          // [128,32]
  float* xa1  = ws + 266240;          // [128,32]
  float* aS   = ws + 270336;          // [<=4096,16] src-table
  float* aD   = ws + 335872;          // [<=4096,16] dst-table
  float* Sv   = ws + 401408;          // [32]
  float* eupT = ws + 401440;          // [128,4096] transposed uplink col0
  float* ednT = ws + 925728;          // [4096,128] transposed downlink col0
  float* tmp  = aS;                   // [4096,32] alias (free at conv1 start)

  float* out_f = (float*)d_out;

  // One-time pack+transpose of edge scalars (coalesced reuse in 3 convs).
  k_pack_transpose<<<CDIV(N_UE*N_AP,256),256,0,stream>>>(eup, eupT, N_UE, N_AP);
  k_pack_transpose<<<CDIV(N_UE*N_AP,256),256,0,stream>>>(edn, ednT, N_AP, N_UE);

  // ---------------- conv1 (Ue2Ap) ----------------
  k_node_mlp2<<<CDIV(N_UE,64),64,0,stream>>>(x_ue,1,1, PF(c1_msg_ue),PF(c1_msg_ue+1),
                                             PF(c1_msg_ue+2),PF(c1_msg_ue+3),
                                             32, tmp,32, N_UE);
  k_colsum<<<32,256,0,stream>>>(tmp, N_UE, 32, Sv);
  k_bcast<<<CDIV(N_AP*32,64),64,0,stream>>>(xa0, Sv, N_AP*32, 32);
  k_edge_msg<<<dim3(N_AP,8),32,0,stream>>>(eupT, nullptr, nullptr,
      PF(c1_edge_up), PF(c1_edge_up+1), PF(c1_edge_up+2), PF(c1_edge_up+3),
      xa0, 32, 32, N_UE, N_UE/8);
  k_copy_strided<<<CDIV(N_UE,64),64,0,stream>>>(xu1,32, x_ue,1, N_UE);
  k_node_mlp2<<<CDIV(N_UE,64),64,0,stream>>>(x_ue,1,1, PF(c1_upd_ue),PF(c1_upd_ue+1),
                                             PF(c1_upd_ue+2),PF(c1_upd_ue+3),
                                             31, xu1+1,32, N_UE);
  k_edge_msg<<<dim3(N_UE,1),32,0,stream>>>(ednT, nullptr, nullptr,
      PF(c1_edge_dn), PF(c1_edge_dn+1), PF(c1_edge_dn+2), PF(c1_edge_dn+3),
      xu1+1, 31, 32, N_AP, N_AP);

  // ---------------- conv2 / conv3 (PowerConv_wAP) ----------------
  auto powerconv = [&](const float* xu_o, const float* xa_o,
                       float* xu_n, float* xa_n,
                       int pmu, int pma, int puu, int pua) {
    // uplink: msg_ue over (ue -> ap)
    const float* W1u = PF(pmu);
    k_lin_partial<<<CDIV(N_UE*16,256),256,0,stream>>>(xu_o,32,32, W1u,        aS, N_UE);
    k_lin_partial<<<CDIV(N_AP*16,256),256,0,stream>>>(xa_o,32,32, W1u+33*16,  aD, N_AP);
    k_node_mlp2<<<CDIV(N_AP,64),64,0,stream>>>(xa_o,32,32, PF(pua),PF(pua+1),
                                               PF(pua+2),PF(pua+3), 32, xa_n,32, N_AP);
    k_edge_msg<<<dim3(N_AP,8),32,0,stream>>>(eupT, aS, aD, W1u+32*16, PF(pmu+1),
                                             PF(pmu+2), PF(pmu+3),
                                             xa_n, 32, 32, N_UE, N_UE/8);
    // downlink: msg_ap over (ap -> ue)
    const float* W1d = PF(pma);
    k_lin_partial<<<CDIV(N_AP*16,256),256,0,stream>>>(xa_o,32,32, W1d,        aS, N_AP);
    k_lin_partial<<<CDIV(N_UE*16,256),256,0,stream>>>(xu_o,32,32, W1d+33*16,  aD, N_UE);
    k_copy_strided<<<CDIV(N_UE,64),64,0,stream>>>(xu_n,32, xu_o,32, N_UE);
    k_node_mlp2<<<CDIV(N_UE,64),64,0,stream>>>(xu_o,32,32, PF(puu),PF(puu+1),
                                               PF(puu+2),PF(puu+3), 31, xu_n+1,32, N_UE);
    k_edge_msg<<<dim3(N_UE,1),32,0,stream>>>(ednT, aS, aD, W1d+32*16, PF(pma+1),
                                             PF(pma+2), PF(pma+3),
                                             xu_n+1, 31, 32, N_AP, N_AP);
  };

  powerconv(xu1, xa0, xu0, xa1, c2_msg_ue, c2_msg_ap, c2_upd_ue, c2_upd_ap);
  powerconv(xu0, xa1, xu1, xa0, c3_msg_ue, c3_msg_ap, c3_upd_ue, c3_upd_ap);

  // ---------------- power head + outputs ----------------
  k_power<<<CDIV(N_UE,64),64,0,stream>>>(xu1, PF(power1), PF(power1+1),
                                         PF(power2), PF(power2+1), out_f, N_UE);
  k_copy<<<CDIV(N_AP*32,64),64,0,stream>>>(out_f + 2*N_UE, xa0, N_AP*32);
#undef PF
}
